// MultiHeadAttention_17171279250119
// MI455X (gfx1250) — compile-verified, matrix operand order checked by reading, not run on hardware
//
#include <hip/hip_runtime.h>
#include <math.h>

typedef __attribute__((ext_vector_type(16))) _Float16 v16h;
typedef __attribute__((ext_vector_type(8)))  _Float16 v8h;
typedef __attribute__((ext_vector_type(8)))  float    v8f;

constexpr int   B  = 4, S = 2048, D = 1024, H = 16, DH = 64;
constexpr int   BS = B * S;
constexpr float INV_TEMP = 0.125f;      // 1/sqrt(DH)
constexpr float EPS = 1e-6f;

// ---------------- workspace layout (in halves) ----------------
constexpr size_t OFF_WQ  = 0;
constexpr size_t OFF_WK  = OFF_WQ  + (size_t)D * D;
constexpr size_t OFF_WV  = OFF_WK  + (size_t)D * D;
constexpr size_t OFF_WFC = OFF_WV  + (size_t)D * D;
constexpr size_t OFF_QH  = OFF_WFC + (size_t)D * D;   // [B,H,S,DH] f16, pre-scaled 1/TEMP
constexpr size_t OFF_KH  = OFF_QH  + (size_t)BS * D;  // [B,H,S,DH] f16
constexpr size_t OFF_VT  = OFF_KH  + (size_t)BS * D;  // [B,H,DH,S] f16 (transposed)
constexpr size_t OFF_O16 = OFF_VT  + (size_t)BS * D;  // [BS,D]     f16 merged heads
constexpr size_t OFF_END = OFF_O16 + (size_t)BS * D;  // then f32 y[BS,D]

static __device__ __forceinline__ v8f wmma_f16(v16h a, v16h b, v8f c) {
  return __builtin_amdgcn_wmma_f32_16x16x32_f16(false, a, false, b, (short)0, c,
                                                false, false);
}

// A/B fragment loader, f16 row-major source. Tile origin = base, row stride =
// `stride` halves. ISA 16-bit layout: lanes 0-15 row r hold k {kb..kb+7,
// kb+16..kb+23} with kb=0; lanes 16-31 same rows with kb=8.
static __device__ __forceinline__ v16h frag_h(const _Float16* base, int stride,
                                              int lane) {
  int r  = lane & 15;
  int kb = (lane >> 4) << 3;
  const _Float16* p = base + (size_t)r * stride + kb;
  union { v16h v; v8h h[2]; } u;
  u.h[0] = *(const v8h*)(p);
  u.h[1] = *(const v8h*)(p + 16);
  return u.v;
}

// Same fragment but from an f32 source, converting on the fly.
static __device__ __forceinline__ v16h frag_f32(const float* base, int stride,
                                                int lane) {
  int r  = lane & 15;
  int kb = (lane >> 4) << 3;
  const float* p = base + (size_t)r * stride + kb;
  v16h v;
#pragma unroll
  for (int i = 0; i < 8; ++i) v[i] = (_Float16)p[i];
#pragma unroll
  for (int i = 0; i < 8; ++i) v[8 + i] = (_Float16)p[16 + i];
  return v;
}

// ---------------- kernel 1: weight f32 -> f16 ----------------
__global__ void __launch_bounds__(256) cvt_kernel(const float* __restrict__ src,
                                                  _Float16* __restrict__ dst,
                                                  int n) {
  int i = blockIdx.x * 256 + threadIdx.x;
  if (i < n) dst[i] = (_Float16)src[i];
}

// ---------------- kernel 2: QKV projection -------------------
// One 16x16 output tile per wave, K=1024.  Q gets 1/TEMP folded in.
// Q,K stored [b,h,s,dh]; V stored transposed [b,h,dh,s].
__global__ void __launch_bounds__(256)
proj_qkv_kernel(const float* __restrict__ qin, const float* __restrict__ kin,
                const float* __restrict__ vin, const _Float16* __restrict__ wsh,
                _Float16* __restrict__ qh, _Float16* __restrict__ kh,
                _Float16* __restrict__ vT) {
  constexpr int TILES = (BS / 16) * (D / 16);  // 32768 per projection
  int lane = threadIdx.x & 31;
  int wid  = blockIdx.x * 8 + (threadIdx.x >> 5);
  int proj = wid / TILES;
  int rem  = wid % TILES;
  int mt   = rem >> 6;          // token tile (BS/16)
  int nt   = rem & 63;          // feature tile (D/16)

  const float*    x = (proj == 0) ? qin : (proj == 1) ? kin : vin;
  const _Float16* w = wsh + ((proj == 0) ? OFF_WQ : (proj == 1) ? OFF_WK : OFF_WV);
  const float*    ab = x + (size_t)(mt * 16) * D;
  const _Float16* bb = w + (size_t)(nt * 16) * D;

  v8f c = {};
  for (int k0 = 0; k0 < D; k0 += 32) {
    __builtin_prefetch(bb + k0 + 64, 0, 0);
    v16h a = frag_f32(ab + k0, D, lane);   // y = x @ W^T : B(k,n) = W[n,k]
    v16h b = frag_h(bb + k0, D, lane);
    c = wmma_f16(a, b, c);
  }

  float scale = (proj == 0) ? INV_TEMP : 1.0f;
  int n  = nt * 16 + (lane & 15);
  int hh = n >> 6, dh = n & 63;
#pragma unroll
  for (int r = 0; r < 8; ++r) {
    int tok = mt * 16 + r + ((lane >> 4) << 3);
    int bb_ = tok >> 11;          // / S
    int ss  = tok & (S - 1);
    size_t bh = (size_t)bb_ * H + hh;
    _Float16 hv = (_Float16)(c[r] * scale);
    if (proj == 2)      vT[(bh * DH + dh) * S + ss] = hv;
    else if (proj == 0) qh[(bh * S + ss) * DH + dh] = hv;
    else                kh[(bh * S + ss) * DH + dh] = hv;
  }
}

// ---------------- kernel 3: fused attention ------------------
// Block = (query-tile qt, head bh).  Full 16x2048 score strip in LDS:
// scores f32 (128KB) + exp probs f16 (64KB) < 320KB CDNA5 WGP LDS.
// attn probs written to HBM exactly once; P.V runs out of LDS.
__global__ void __launch_bounds__(256)
attn_kernel(const _Float16* __restrict__ qh, const _Float16* __restrict__ kh,
            const _Float16* __restrict__ vT, _Float16* __restrict__ o16,
            float* __restrict__ attn_out) {
  __shared__ float    sc[16][S];
  __shared__ _Float16 pr[16][S];
  __shared__ float    part[16][16];
  __shared__ float    rowmax[16], rowsum[16];

  int qt = blockIdx.x;            // 0..127
  int bh = blockIdx.y;            // 0..63
  int lane = threadIdx.x & 31, wave = threadIdx.x >> 5;

  // --- scores = (Q/TEMP) . K^T, 8 waves stripe the 128 key tiles ---
  const _Float16* qb = qh + ((size_t)bh * S + qt * 16) * DH;
  v16h qa0 = frag_h(qb + 0, DH, lane);
  v16h qa1 = frag_h(qb + 32, DH, lane);
  for (int kt = wave; kt < S / 16; kt += 8) {
    const _Float16* kb = kh + ((size_t)bh * S + kt * 16) * DH;
    v8f c = {};
    c = wmma_f16(qa0, frag_h(kb + 0, DH, lane), c);
    c = wmma_f16(qa1, frag_h(kb + 32, DH, lane), c);
    int n = kt * 16 + (lane & 15);
#pragma unroll
    for (int r = 0; r < 8; ++r) sc[r + ((lane >> 4) << 3)][n] = c[r];
  }
  __syncthreads();

  // --- softmax across 2048 keys; 16 lanes cooperate per row ---
  int row = threadIdx.x >> 4, wc = threadIdx.x & 15;
  float lmax = -3.0e38f;
  for (int j = 0; j < S / 16; ++j) lmax = fmaxf(lmax, sc[row][wc + 16 * j]);
  part[row][wc] = lmax;
  __syncthreads();
  if (wc == 0) {
    float m = part[row][0];
#pragma unroll
    for (int i = 1; i < 16; ++i) m = fmaxf(m, part[row][i]);
    rowmax[row] = m;
  }
  __syncthreads();
  float rmax = rowmax[row], lsum = 0.f;
  for (int j = 0; j < S / 16; ++j) {
    int cx = wc + 16 * j;
    float e = __expf(sc[row][cx] - rmax);
    sc[row][cx] = e;
    pr[row][cx] = (_Float16)e;        // unnormalized; 1/sum folded into O
    lsum += e;
  }
  part[row][wc] = lsum;
  __syncthreads();
  if (wc == 0) {
    float s = 0.f;
#pragma unroll
    for (int i = 0; i < 16; ++i) s += part[row][i];
    rowsum[row] = s;
  }
  __syncthreads();

  // --- single HBM write of normalized probabilities ---
  float inv = 1.0f / rowsum[row];
  float* ar = attn_out + ((size_t)bh * S + qt * 16 + row) * S;
  for (int j = 0; j < S / 16; ++j) {
    int cx = wc + 16 * j;
    ar[cx] = sc[row][cx] * inv;
  }
  __syncthreads();

  // --- O = P.V straight from LDS; waves 0..3 own the 4 dh tiles ---
  if (wave < 4) {
    const _Float16* vb = vT + ((size_t)bh * DH + wave * 16) * S;
    int am = lane & 15, akb = (lane >> 4) << 3;
    v8f c = {};
    for (int k0 = 0; k0 < S; k0 += 32) {
      union { v16h v; v8h h[2]; } ua;
      ua.h[0] = *(const v8h*)(&pr[am][k0 + akb]);
      ua.h[1] = *(const v8h*)(&pr[am][k0 + akb + 16]);
      v16h b = frag_h(vb + k0, S, lane);   // B(k=key, n=dh) contiguous in vT
      c = wmma_f16(ua.v, b, c);
    }
    int hh = bh & (H - 1), bsel = bh >> 4;
    int dh = wave * 16 + (lane & 15);
#pragma unroll
    for (int r = 0; r < 8; ++r) {
      int m = r + ((lane >> 4) << 3);
      float val = c[r] * (1.0f / rowsum[m]);
      size_t tok = (size_t)bsel * S + qt * 16 + m;
      o16[tok * D + hh * DH + dh] = (_Float16)val;
    }
  }
}

// ---------------- kernel 4: output projection ----------------
__global__ void __launch_bounds__(256)
fc_kernel(const _Float16* __restrict__ o16, const _Float16* __restrict__ wfc,
          float* __restrict__ y) {
  int lane = threadIdx.x & 31;
  int wid  = blockIdx.x * 8 + (threadIdx.x >> 5);
  int mt = wid >> 6, nt = wid & 63;
  const _Float16* ab = o16 + (size_t)(mt * 16) * D;
  const _Float16* bb = wfc + (size_t)(nt * 16) * D;
  v8f c = {};
  for (int k0 = 0; k0 < D; k0 += 32) {
    __builtin_prefetch(bb + k0 + 64, 0, 0);
    c = wmma_f16(frag_h(ab + k0, D, lane), frag_h(bb + k0, D, lane), c);
  }
  int col = nt * 16 + (lane & 15);
#pragma unroll
  for (int r = 0; r < 8; ++r) {
    int tok = mt * 16 + r + ((lane >> 4) << 3);
    y[(size_t)tok * D + col] = c[r];
  }
}

// ---------------- kernel 5: residual + layernorm -------------
__global__ void __launch_bounds__(256)
ln_kernel(const float* __restrict__ y, const float* __restrict__ qin,
          const float* __restrict__ lnw, const float* __restrict__ lnb,
          float* __restrict__ out) {
  __shared__ float red[256];
  int row = blockIdx.x, tid = threadIdx.x;
  float t[4], sum = 0.f, sq = 0.f;
#pragma unroll
  for (int j = 0; j < 4; ++j) {
    int col = tid + 256 * j;
    float v = y[(size_t)row * D + col] + qin[(size_t)row * D + col];
    t[j] = v; sum += v; sq += v * v;
  }
  red[tid] = sum; __syncthreads();
  for (int s = 128; s > 0; s >>= 1) { if (tid < s) red[tid] += red[tid + s]; __syncthreads(); }
  float mu = red[0] / D; __syncthreads();
  red[tid] = sq; __syncthreads();
  for (int s = 128; s > 0; s >>= 1) { if (tid < s) red[tid] += red[tid + s]; __syncthreads(); }
  float var = red[0] / D - mu * mu;
  float rs = rsqrtf(var + EPS);
#pragma unroll
  for (int j = 0; j < 4; ++j) {
    int col = tid + 256 * j;
    out[(size_t)row * D + col] = (t[j] - mu) * rs * lnw[col] + lnb[col];
  }
}

// ---------------- host launcher ------------------------------
extern "C" void kernel_launch(void* const* d_in, const int* in_sizes, int n_in,
                              void* d_out, int out_size, void* d_ws,
                              size_t ws_size, hipStream_t stream) {
  const float* qin = (const float*)d_in[0];
  const float* kin = (const float*)d_in[1];
  const float* vin = (const float*)d_in[2];
  const float* wq  = (const float*)d_in[3];
  const float* wk  = (const float*)d_in[4];
  const float* wv  = (const float*)d_in[5];
  const float* wfc = (const float*)d_in[6];
  const float* lnw = (const float*)d_in[7];
  const float* lnb = (const float*)d_in[8];

  _Float16* wsh = (_Float16*)d_ws;
  float*    yf  = (float*)((char*)d_ws + OFF_END * sizeof(_Float16));
  float* out_o    = (float*)d_out;                    // [B,S,D]
  float* out_attn = out_o + (size_t)BS * D;           // [B,H,S,S]

  // 1) weights f32 -> f16 (one-time; reused 512x by GEMMs)
  const int WN = D * D, WBLK = (WN + 255) / 256;
  cvt_kernel<<<WBLK, 256, 0, stream>>>(wq,  wsh + OFF_WQ,  WN);
  cvt_kernel<<<WBLK, 256, 0, stream>>>(wk,  wsh + OFF_WK,  WN);
  cvt_kernel<<<WBLK, 256, 0, stream>>>(wv,  wsh + OFF_WV,  WN);
  cvt_kernel<<<WBLK, 256, 0, stream>>>(wfc, wsh + OFF_WFC, WN);

  // 2) QKV projections: 3 * 512 * 64 tiles / 8 waves per block
  proj_qkv_kernel<<<3 * (BS / 16) * (D / 16) / 8, 256, 0, stream>>>(
      qin, kin, vin, wsh, wsh + OFF_QH, wsh + OFF_KH, wsh + OFF_VT);

  // 3) fused attention: (128 query tiles) x (64 batch*head)
  attn_kernel<<<dim3(S / 16, B * H), 256, 0, stream>>>(
      wsh + OFF_QH, wsh + OFF_KH, wsh + OFF_VT, wsh + OFF_O16, out_attn);

  // 4) output projection
  fc_kernel<<<(BS / 16) * (D / 16) / 8, 256, 0, stream>>>(
      wsh + OFF_O16, wsh + OFF_WFC, yf);

  // 5) residual + layernorm
  ln_kernel<<<BS, 256, 0, stream>>>(yf, qin, lnw, lnb, out_o);
}
